// SimpleGraphTransformer_33303176413376
// MI455X (gfx1250) — matrix operand order repaired and checked by reading, not compiled, not run on hardware
//
#include <hip/hip_runtime.h>

#define D     128
#define HEADS 4
#define CH    32

typedef __bf16 bf16_t;
typedef bf16_t v16bf __attribute__((ext_vector_type(16)));
typedef bf16_t v8bf  __attribute__((ext_vector_type(8)));
typedef float  v8f   __attribute__((ext_vector_type(8)));

// ---------------- utility kernels ----------------

__global__ void zero_f32_kernel(float* __restrict__ p, int n) {
    int i = blockIdx.x * blockDim.x + threadIdx.x;
    if (i < n) p[i] = 0.0f;
}

__global__ void zero_u32_kernel(unsigned int* __restrict__ p, int n) {
    int i = blockIdx.x * blockDim.x + threadIdx.x;
    if (i < n) p[i] = 0u;   // below enc() of any finite float
}

__global__ void cast_f32_bf16_kernel(const float* __restrict__ in,
                                     bf16_t* __restrict__ out, int n) {
    int i = blockIdx.x * blockDim.x + threadIdx.x;
    if (i < n) out[i] = (bf16_t)in[i];
}

// Transpose+cast the four weight matrices: wt[w][n][k] = bf16(W_w[k][n])
__global__ void prep_weights_kernel(const float* __restrict__ Wq,
                                    const float* __restrict__ Wk,
                                    const float* __restrict__ Wv,
                                    const float* __restrict__ Ws,
                                    bf16_t* __restrict__ wt) {
    int i = blockIdx.x * blockDim.x + threadIdx.x;
    if (i >= D * D) return;
    int w = blockIdx.y;
    const float* W = (w == 0) ? Wq : (w == 1) ? Wk : (w == 2) ? Wv : Ws;
    int k = i / D, n = i % D;
    wt[(size_t)w * D * D + (size_t)n * D + k] = (bf16_t)W[(size_t)k * D + n];
}

// ---------------- bf16 WMMA GEMM: out[N,128] = x[N,128] @ W[128,128] + bias ----
// One wave computes a 16x128 row-block (8 WMMA accumulators), K looped 4x32.
// A fragment (16-bit 16x32, ISA 7.12.2): lane half 0 -> K {ks..ks+7, ks+16..ks+23},
// lane half 1 -> +8.  B fragment (32x16): same K order as A,
// from the pre-transposed wt (wt[n][k]).
// All 4 A fragments are preloaded; all 8 B fragments of a K-step are batch-loaded
// into a register array before the WMMA chain so the loads pipeline (partial
// s_wait_loadcnt) instead of a full wait before every WMMA.

__global__ __launch_bounds__(256) void gemm_bf16_wmma_kernel(
    const bf16_t* __restrict__ xbf,   // [nrows,128]
    const bf16_t* __restrict__ wt,    // [128(n),128(k)]  (= W transposed)
    const float*  __restrict__ bias,  // [128]
    float* __restrict__ out,          // [nrows,128]
    int nrows)
{
    const int lane  = threadIdx.x & 31;
    const int wave  = threadIdx.x >> 5;
    const int mtile = blockIdx.x * 8 + wave;
    if (mtile * 16 >= nrows) return;

    const int half = lane >> 4;       // which 16-lane half
    const int l16  = lane & 15;

    int arow = mtile * 16 + l16;
    if (arow >= nrows) arow = nrows - 1;           // clamp (N%16==0 normally)
    const bf16_t* xr = xbf + (size_t)arow * D;
    const bf16_t* wr = wt + (size_t)l16 * D;       // lane's B row base (col = t*16+l16)

    // preload all A fragments (whole 128-wide row contribution for this lane)
    v16bf areg[4];
    #pragma unroll
    for (int i = 0; i < 4; ++i) {
        const int kA = i * 32 + half * 8;
        v8bf alo = *(const v8bf*)(xr + kA);
        v8bf ahi = *(const v8bf*)(xr + kA + 16);
        areg[i]  = __builtin_shufflevector(alo, ahi,
                        0,1,2,3,4,5,6,7,8,9,10,11,12,13,14,15);
    }

    v8f acc[8] = {};

    #pragma unroll
    for (int i = 0; i < 4; ++i) {
        const int kB = i * 32 + half * 8;
        // batch-load all 8 B fragments for this K-step
        v16bf breg[8];
        #pragma unroll
        for (int t = 0; t < 8; ++t) {
            v8bf blo = *(const v8bf*)(wr + (size_t)t * 16 * D + kB);
            v8bf bhi = *(const v8bf*)(wr + (size_t)t * 16 * D + kB + 16);
            breg[t] = __builtin_shufflevector(blo, bhi,
                        0,1,2,3,4,5,6,7,8,9,10,11,12,13,14,15);
        }
        // WMMA chain (same A operand across all 8)
        #pragma unroll
        for (int t = 0; t < 8; ++t)
            acc[t] = __builtin_amdgcn_wmma_f32_16x16x32_bf16(
                         false, areg[i], false, breg[t], (short)0, acc[t],
                         false, false);
    }

    // C layout: element r -> row mtile*16 + r + half*8, col = ntile*16 + l16
    const int row0 = mtile * 16 + half * 8;
    #pragma unroll
    for (int t = 0; t < 8; ++t) {
        const int col  = t * 16 + l16;
        const float bc = bias[col];
        #pragma unroll
        for (int r = 0; r < 8; ++r) {
            const int row = row0 + r;
            if (row < nrows)
                out[(size_t)row * D + col] = acc[t][r] + bc;
        }
    }
}

// ---------------- edge kernels ----------------

// One wave per (edge, head): alpha = q[dst].k[src]/sqrt(32); atomicMax segment max.
__global__ __launch_bounds__(256) void edge_alpha_kernel(
    const float* __restrict__ q, const float* __restrict__ k,
    const int* __restrict__ src, const int* __restrict__ dst,
    float* __restrict__ alpha, unsigned int* __restrict__ amax_enc, int E)
{
    const int w    = blockIdx.x * 8 + (threadIdx.x >> 5);
    const int lane = threadIdx.x & 31;
    const int e = w >> 2;
    const int h = w & 3;
    if (e >= E) return;
    const int s = src[e], d = dst[e];
    float p = q[(size_t)d * D + h * CH + lane] * k[(size_t)s * D + h * CH + lane];
    #pragma unroll
    for (int off = 16; off > 0; off >>= 1) p += __shfl_xor(p, off, 32);
    if (lane == 0) {
        const float a = p * 0.17677669529663687f;   // 1/sqrt(32)
        alpha[(size_t)e * HEADS + h] = a;
        unsigned int u = __float_as_uint(a);
        unsigned int enc = u ^ ((unsigned int)((int)u >> 31) | 0x80000000u);
        atomicMax(&amax_enc[(size_t)d * HEADS + h], enc);
    }
}

// ea = exp(alpha - amax[dst]); alpha <- ea; denom[dst] += ea
__global__ void edge_exp_kernel(const int* __restrict__ dst,
                                float* __restrict__ alpha,
                                const unsigned int* __restrict__ amax_enc,
                                float* __restrict__ denom, int E)
{
    int i = blockIdx.x * blockDim.x + threadIdx.x;
    if (i >= E * HEADS) return;
    const int e = i >> 2, h = i & 3;
    const int d = dst[e];
    const unsigned int enc = amax_enc[(size_t)d * HEADS + h];
    const unsigned int u = (enc & 0x80000000u) ? (enc ^ 0x80000000u) : ~enc;
    const float m  = __uint_as_float(u);
    const float ea = __expf(alpha[i] - m);
    alpha[i] = ea;
    atomicAdd(&denom[(size_t)d * HEADS + h], ea);
}

// acc[dst, c] += ea[e,h(c)] * v[src, c]   (normalization deferred to finalize)
__global__ void edge_scatter_kernel(const int* __restrict__ src,
                                    const int* __restrict__ dst,
                                    const float* __restrict__ alpha,
                                    const float* __restrict__ v,
                                    float* __restrict__ acc, int E)
{
    int i = blockIdx.x * blockDim.x + threadIdx.x;
    if (i >= E * D) return;
    const int e = i >> 7;
    const int c = i & 127;
    const int h = c >> 5;
    const float val = alpha[(size_t)e * HEADS + h] * v[(size_t)src[e] * D + c];
    atomicAdd(&acc[(size_t)dst[e] * D + c], val);
}

// out = relu(acc/max(denom,1e-16) + skip)
__global__ void finalize_kernel(const float* __restrict__ acc,
                                const float* __restrict__ denom,
                                const float* __restrict__ skip,
                                float* __restrict__ out, int n)
{
    int i = blockIdx.x * blockDim.x + threadIdx.x;
    if (i >= n) return;
    const int node = i >> 7;
    const int h    = (i & 127) >> 5;
    const float den = fmaxf(denom[(size_t)node * HEADS + h], 1e-16f);
    const float o   = acc[i] / den + skip[i];
    out[i] = fmaxf(o, 0.0f);
}

// ---------------- host launcher ----------------

extern "C" void kernel_launch(void* const* d_in, const int* in_sizes, int n_in,
                              void* d_out, int out_size, void* d_ws, size_t ws_size,
                              hipStream_t stream)
{
    const float* x   = (const float*)d_in[0];
    const int* eidx  = (const int*)d_in[1];
    const int N = in_sizes[0] / D;
    const int E = in_sizes[1] / 2;
    const int* src = eidx;
    const int* dst = eidx + E;

    const float* W[2][4];
    const float* B[2][4];
    for (int l = 0; l < 2; ++l)
        for (int j = 0; j < 4; ++j) {
            W[l][j] = (const float*)d_in[2 + l * 8 + j];
            B[l][j] = (const float*)d_in[6 + l * 8 + j];
        }

    // workspace carve-out
    char*  ws  = (char*)d_ws;
    size_t off = 0;
    auto carve = [&](size_t bytes) -> void* {
        void* p = ws + off;
        off = (off + bytes + 255) & ~(size_t)255;
        return p;
    };
    bf16_t*       xbf   = (bf16_t*)      carve((size_t)N * D * sizeof(bf16_t));
    bf16_t*       wt    = (bf16_t*)      carve((size_t)4 * D * D * sizeof(bf16_t));
    float*        qb    = (float*)       carve((size_t)N * D * sizeof(float));
    float*        kb    = (float*)       carve((size_t)N * D * sizeof(float));
    float*        vb    = (float*)       carve((size_t)N * D * sizeof(float));
    float*        sb    = (float*)       carve((size_t)N * D * sizeof(float));
    float*        alpha = (float*)       carve((size_t)E * HEADS * sizeof(float));
    unsigned int* amax  = (unsigned int*)carve((size_t)N * HEADS * sizeof(unsigned int));
    float*        denom = (float*)       carve((size_t)N * HEADS * sizeof(float));
    float*        acc   = (float*)       carve((size_t)N * D * sizeof(float));
    float*        h1    = (float*)       carve((size_t)N * D * sizeof(float));

    const int ND  = N * D;
    const int NH  = N * HEADS;
    const int EH  = E * HEADS;
    const int ED  = E * D;
    const int TPB = 256;

    for (int l = 0; l < 2; ++l) {
        const float* fin  = (l == 0) ? x  : h1;
        float*       fout = (l == 0) ? h1 : (float*)d_out;

        // cast features, prep (transpose+cast) weights
        cast_f32_bf16_kernel<<<(ND + TPB - 1) / TPB, TPB, 0, stream>>>(fin, xbf, ND);
        dim3 gw((D * D + TPB - 1) / TPB, 4);
        prep_weights_kernel<<<gw, TPB, 0, stream>>>(W[l][0], W[l][1], W[l][2], W[l][3], wt);

        // per-call init of accumulators (workspace is not re-poisoned by harness)
        zero_f32_kernel<<<(ND + TPB - 1) / TPB, TPB, 0, stream>>>(acc, ND);
        zero_f32_kernel<<<(NH + TPB - 1) / TPB, TPB, 0, stream>>>(denom, NH);
        zero_u32_kernel<<<(NH + TPB - 1) / TPB, TPB, 0, stream>>>(amax, NH);

        // four WMMA GEMMs: q, k, v, skip
        const int mtiles = (N + 15) / 16;
        const int gblk   = (mtiles + 7) / 8;
        gemm_bf16_wmma_kernel<<<gblk, TPB, 0, stream>>>(xbf, wt + 0 * D * D, B[l][0], qb, N);
        gemm_bf16_wmma_kernel<<<gblk, TPB, 0, stream>>>(xbf, wt + 1 * D * D, B[l][1], kb, N);
        gemm_bf16_wmma_kernel<<<gblk, TPB, 0, stream>>>(xbf, wt + 2 * D * D, B[l][2], vb, N);
        gemm_bf16_wmma_kernel<<<gblk, TPB, 0, stream>>>(xbf, wt + 3 * D * D, B[l][3], sb, N);

        // attention: logits + segment max
        edge_alpha_kernel<<<(EH + 7) / 8, TPB, 0, stream>>>(qb, kb, src, dst, alpha, amax, E);
        // exp + segment sum
        edge_exp_kernel<<<(EH + TPB - 1) / TPB, TPB, 0, stream>>>(dst, alpha, amax, denom, E);
        // weighted scatter-add of values (unnormalized)
        edge_scatter_kernel<<<(ED + TPB - 1) / TPB, TPB, 0, stream>>>(src, dst, alpha, vb, acc, E);
        // normalize + skip + relu
        finalize_kernel<<<(ND + TPB - 1) / TPB, TPB, 0, stream>>>(acc, denom, sb, fout, ND);
    }
}
